// ResBlock2_77129022701585
// MI455X (gfx1250) — matrix operand order repaired and checked by reading, not compiled, not run on hardware
//
#include <hip/hip_runtime.h>
#include <hip/hip_bf16.h>

// ---------------------------------------------------------------------------
// CDNA5 (gfx1250) wave32 WMMA implementation of the dual-branch gated
// attention block.  All heavy GEMMs use v_wmma_f32_16x16x32_bf16.
// Flash kernel uses 32-row register blocking per wave so every B-fragment
// feeds two WMMAs (halves operand traffic per matrix op).
// ---------------------------------------------------------------------------

typedef __bf16 bf16_t;
typedef __attribute__((ext_vector_type(16))) __bf16 v16bf;
typedef __attribute__((ext_vector_type(8)))  __bf16 v8bf;
typedef __attribute__((ext_vector_type(8)))  float   v8f;

union VF16 { v16bf v; v8bf h[2]; };

#define N_TOK  4096
#define D_IN   512
#define D_HID  256
#define D_BOX  22
#define D_BOXP 32     // box dim padded to 32 so one 16x16x32 k-step covers it

#define BM  64        // rows of O per block (flash kernel)
#define BN  64        // score-column block width
#define SLD 72        // padded LDS row stride (elements) for the S tile

// ---- WMMA fragment loaders (layouts per CDNA5 ISA 7.12.2) ------

// A matrix 16x32 bf16: lane L holds row M=L&15; elements 0..7 = K kh*8..+7,
// elements 8..15 = K 16+kh*8..+7  (kh = L>>4).
__device__ __forceinline__ v16bf load_a_frag(const bf16_t* __restrict__ src,
                                             int row0, int ld, int k0) {
  const int l  = threadIdx.x & 31;
  const int m  = l & 15;
  const int kh = l >> 4;
  const bf16_t* p = src + (size_t)(row0 + m) * ld + k0 + kh * 8;
  VF16 r;
  r.h[0] = *(const v8bf*)(p);
  r.h[1] = *(const v8bf*)(p + 16);
  return r.v;
}

// B matrix 32x16 bf16: lane L holds column N=L&15; elements 0..7 = K kh*8..+7,
// elements 8..15 = K 16+kh*8..+7.  src is "row per output column" (e.g. transposed
// weights, Q rows, G rows, transposed V) so the K values are contiguous.
__device__ __forceinline__ v16bf load_b_frag(const bf16_t* __restrict__ src,
                                             int col0, int ld, int k0) {
  const int l  = threadIdx.x & 31;
  const int n  = l & 15;
  const int kh = l >> 4;
  const bf16_t* p = src + (size_t)(col0 + n) * ld + k0 + kh * 8;
  VF16 r;
  r.h[0] = *(const v8bf*)(p);
  r.h[1] = *(const v8bf*)(p + 16);
  return r.v;
}

__device__ __forceinline__ v8f wmma_bf16(v16bf a, v16bf b, v8f c) {
  return __builtin_amdgcn_wmma_f32_16x16x32_bf16(
      /*neg_a=*/false, a, /*neg_b=*/false, b,
      /*c_mod=*/(short)0, c, /*reuse_a=*/false, /*reuse_b=*/false);
}

// ---- small conversion / transpose kernels ---------------------------------

__global__ void cvt_f32_to_bf16_kernel(const float* __restrict__ in,
                                       bf16_t* __restrict__ out, int n) {
  int i = blockIdx.x * blockDim.x + threadIdx.x;
  if (i < n) out[i] = (bf16_t)in[i];
}

__global__ void cvt_pad_box_kernel(const float* __restrict__ in,
                                   bf16_t* __restrict__ out) {
  int i = blockIdx.x * blockDim.x + threadIdx.x;
  if (i >= N_TOK * D_BOXP) return;
  int r = i / D_BOXP, c = i % D_BOXP;
  out[i] = (c < D_BOX) ? (bf16_t)in[r * D_BOX + c] : (bf16_t)0.0f;
}

// W is [in_dim][out_dim] fp32 -> Wt [out_dim][pad] bf16 (zero padded)
__global__ void transpose_w_kernel(const float* __restrict__ W,
                                   bf16_t* __restrict__ Wt,
                                   int in_dim, int out_dim, int pad) {
  int i = blockIdx.x * blockDim.x + threadIdx.x;
  if (i >= out_dim * pad) return;
  int o = i / pad, k = i % pad;
  Wt[i] = (k < in_dim) ? (bf16_t)W[(size_t)k * out_dim + o] : (bf16_t)0.0f;
}

// ---- projection GEMM: Out[4096 x 256] = X[4096 x kd] @ W ------------------
// Block = 256 threads (8 waves), 64x64 output tile; wave (wm,wn) does 16x32.
__global__ __launch_bounds__(256)
void proj_kernel(const bf16_t* __restrict__ X, const bf16_t* __restrict__ Wt,
                 bf16_t* __restrict__ Out, int ldin, int kdim,
                 int out_ld, int transposed) {
  const int w    = threadIdx.x >> 5;
  const int wm   = w >> 1;
  const int wn   = w & 1;
  const int row0 = blockIdx.x * 64 + wm * 16;
  const int col0 = blockIdx.y * 64 + wn * 32;

  v8f c0 = {}, c1 = {};
  for (int k0 = 0; k0 < kdim; k0 += 32) {
    v16bf a  = load_a_frag(X, row0, ldin, k0);
    v16bf b0 = load_b_frag(Wt, col0,      ldin, k0);
    v16bf b1 = load_b_frag(Wt, col0 + 16, ldin, k0);
    c0 = wmma_bf16(a, b0, c0);
    c1 = wmma_bf16(a, b1, c1);
  }

  const int l  = threadIdx.x & 31;
  const int n  = l & 15;
  const int kh = l >> 4;
#pragma unroll
  for (int r = 0; r < 8; ++r) {
    int M = row0 + r + 8 * kh;
    if (!transposed) {
      Out[(size_t)M * out_ld + col0 + n]      = (bf16_t)c0[r];
      Out[(size_t)M * out_ld + col0 + 16 + n] = (bf16_t)c1[r];
    } else {  // V stored transposed: Vt[col][row], out_ld = 4096
      Out[(size_t)(col0 + n)      * out_ld + M] = (bf16_t)c0[r];
      Out[(size_t)(col0 + 16 + n) * out_ld + M] = (bf16_t)c1[r];
    }
  }
}

// ---- flash kernel: O_unnorm = S @ V, partial[block] = sum(S) --------------
// 8 waves = (wm in {0,1}: 32-row half) x (wn in {0..3}: col quarter).
// Each wave: scores for rows [wm*32,+32) x cols [wn*16,+16) of each 64-wide
// column block, then O chunk rows [wm*32,+32) x cols [wn*64,+64).
__global__ __launch_bounds__(256)
void flash_kernel(const bf16_t* __restrict__ K1, const bf16_t* __restrict__ Q1,
                  const bf16_t* __restrict__ G1, const bf16_t* __restrict__ Vt1,
                  float* __restrict__ O1,
                  const bf16_t* __restrict__ K2, const bf16_t* __restrict__ Q2,
                  const bf16_t* __restrict__ G2, const bf16_t* __restrict__ Vt2,
                  float* __restrict__ O2,
                  float* __restrict__ partial) {
  const int br = blockIdx.y;
  const bf16_t* Km  = br ? K2  : K1;
  const bf16_t* Qm  = br ? Q2  : Q1;
  const bf16_t* Gm  = br ? G2  : G1;
  const bf16_t* Vtm = br ? Vt2 : Vt1;
  float*        Om  = br ? O2  : O1;

  __shared__ bf16_t Sld[BM * SLD];
  __shared__ float  red[256];

  const int w    = threadIdx.x >> 5;
  const int wm   = w >> 2;                 // 0..1 : 32-row half
  const int wn   = w & 3;                  // 0..3 : 16-col quarter (scores)
  const int row0 = blockIdx.x * BM;
  const int l    = threadIdx.x & 31;
  const int n    = l & 15;
  const int kh   = l >> 4;

  // Hoist K and G A-fragments for both 16-row groups (invariant over j).
  v16bf aK[2][8], aG[2][8];
#pragma unroll
  for (int rg = 0; rg < 2; ++rg) {
    const int arow = row0 + wm * 32 + rg * 16;
#pragma unroll
    for (int kk = 0; kk < 8; ++kk) {
      aK[rg][kk] = load_a_frag(Km, arow, D_HID, kk * 32);
      aG[rg][kk] = load_a_frag(Gm, arow, D_HID, kk * 32);
    }
  }

  v8f o[2][4];
#pragma unroll
  for (int rg = 0; rg < 2; ++rg)
#pragma unroll
    for (int nt = 0; nt < 4; ++nt) o[rg][nt] = {};

  float ssum = 0.0f;

  for (int jb = 0; jb < N_TOK; jb += BN) {
    // Prefetch next column block's operand lines into caches (gfx1250
    // global_prefetch_b8) while this block computes.
    if (jb + BN < N_TOK) {
      const int jn = jb + BN;
      __builtin_prefetch(Qm  + (size_t)(jn + wn * 16 + n) * D_HID + kh * 128, 0, 1);
      __builtin_prefetch(Gm  + (size_t)(jn + wn * 16 + n) * D_HID + kh * 128, 0, 1);
      __builtin_prefetch(Vtm + (size_t)(wn * 64 + kh * 32 + n) * N_TOK + jn, 0, 1);
    }

    // ---- scores: S rows [wm*32,+32), cols [jb+wn*16,+16) -------------------
    const int col = jb + wn * 16;
    v8f ck[2] = {{}, {}};
    v8f cg[2] = {{}, {}};
#pragma unroll
    for (int kk = 0; kk < 8; ++kk) {
      v16bf bq = load_b_frag(Qm, col, D_HID, kk * 32);
      v16bf bg = load_b_frag(Gm, col, D_HID, kk * 32);
#pragma unroll
      for (int rg = 0; rg < 2; ++rg) {
        ck[rg] = wmma_bf16(aK[rg][kk], bq, ck[rg]);
        cg[rg] = wmma_bf16(aG[rg][kk], bg, cg[rg]);
      }
    }
#pragma unroll
    for (int rg = 0; rg < 2; ++rg) {
#pragma unroll
      for (int r = 0; r < 8; ++r) {
        float a = ck[rg][r] * 0.0625f;
        float g = cg[rg][r] * 0.0625f;
        g = g > 0.0f ? g : 0.0f;                // relu gate
        float s = g * __expf(a);                // gated exponential score
        ssum += s;
        Sld[(wm * 32 + rg * 16 + r + 8 * kh) * SLD + wn * 16 + n] = (bf16_t)s;
      }
    }
    __syncthreads();

    // ---- O += S_block @ V_block  (A from LDS, B from transposed V) --------
#pragma unroll
    for (int ks = 0; ks < 2; ++ks) {
      v16bf sa0 = load_a_frag(Sld, wm * 32,      SLD, ks * 32);
      v16bf sa1 = load_a_frag(Sld, wm * 32 + 16, SLD, ks * 32);
#pragma unroll
      for (int nt = 0; nt < 4; ++nt) {
        v16bf bv = load_b_frag(Vtm, wn * 64 + nt * 16, N_TOK, jb + ks * 32);
        o[0][nt] = wmma_bf16(sa0, bv, o[0][nt]);
        o[1][nt] = wmma_bf16(sa1, bv, o[1][nt]);
      }
    }
    __syncthreads();
  }

  // ---- store unnormalized O ------------------------------------------------
#pragma unroll
  for (int rg = 0; rg < 2; ++rg) {
#pragma unroll
    for (int nt = 0; nt < 4; ++nt) {
#pragma unroll
      for (int r = 0; r < 8; ++r) {
        int M = row0 + wm * 32 + rg * 16 + r + 8 * kh;
        Om[(size_t)M * D_HID + wn * 64 + nt * 16 + n] = o[rg][nt][r];
      }
    }
  }

  // ---- deterministic per-block sum(S) -------------------------------------
  red[threadIdx.x] = ssum;
  __syncthreads();
  for (int s = 128; s > 0; s >>= 1) {
    if ((int)threadIdx.x < s) red[threadIdx.x] += red[threadIdx.x + s];
    __syncthreads();
  }
  if (threadIdx.x == 0) partial[br * 64 + blockIdx.x] = red[0];
}

__global__ void reduce_sums_kernel(const float* __restrict__ partial,
                                   float* __restrict__ sums) {
  int b = threadIdx.x;
  if (b < 2) {
    float t = 0.0f;
    for (int i = 0; i < 64; ++i) t += partial[b * 64 + i];  // fixed order
    sums[b] = t;
  }
}

__global__ void finalize_kernel(const float* __restrict__ O1,
                                const float* __restrict__ O2,
                                const float* __restrict__ sums,
                                const float* __restrict__ x,
                                float* __restrict__ out) {
  int i = blockIdx.x * blockDim.x + threadIdx.x;
  if (i >= N_TOK * D_IN) return;
  int r = i >> 9, c = i & 511;
  float v = (c < D_HID) ? O1[r * D_HID + c] / sums[0]
                        : O2[r * D_HID + (c - D_HID)] / sums[1];
  out[i] = v * 0.1f + x[i];
}

// ---------------------------------------------------------------------------

extern "C" void kernel_launch(void* const* d_in, const int* in_sizes, int n_in,
                              void* d_out, int out_size, void* d_ws, size_t ws_size,
                              hipStream_t stream) {
  const float* x   = (const float*)d_in[0];
  const float* box = (const float*)d_in[1];
  const float* WG1 = (const float*)d_in[2];
  const float* WK1 = (const float*)d_in[3];
  const float* WQ1 = (const float*)d_in[4];
  const float* WV1 = (const float*)d_in[5];
  const float* WG2 = (const float*)d_in[6];
  const float* WK2 = (const float*)d_in[7];
  const float* WQ2 = (const float*)d_in[8];
  const float* WV2 = (const float*)d_in[9];
  float* out = (float*)d_out;

  char* ws = (char*)d_ws;
  size_t off = 0;
  auto alloc = [&](size_t bytes) -> void* {
    void* p = ws + off;
    off = (off + bytes + 255) & ~(size_t)255;
    return p;
  };

  bf16_t* xb   = (bf16_t*)alloc((size_t)N_TOK * D_IN   * 2);
  bf16_t* boxb = (bf16_t*)alloc((size_t)N_TOK * D_BOXP * 2);
  bf16_t* WtK1 = (bf16_t*)alloc((size_t)D_HID * D_IN   * 2);
  bf16_t* WtQ1 = (bf16_t*)alloc((size_t)D_HID * D_IN   * 2);
  bf16_t* WtV1 = (bf16_t*)alloc((size_t)D_HID * D_IN   * 2);
  bf16_t* WtK2 = (bf16_t*)alloc((size_t)D_HID * D_IN   * 2);
  bf16_t* WtQ2 = (bf16_t*)alloc((size_t)D_HID * D_IN   * 2);
  bf16_t* WtV2 = (bf16_t*)alloc((size_t)D_HID * D_IN   * 2);
  bf16_t* WtG1 = (bf16_t*)alloc((size_t)D_HID * D_BOXP * 2);
  bf16_t* WtG2 = (bf16_t*)alloc((size_t)D_HID * D_BOXP * 2);
  bf16_t* Kb1  = (bf16_t*)alloc((size_t)N_TOK * D_HID  * 2);
  bf16_t* Qb1  = (bf16_t*)alloc((size_t)N_TOK * D_HID  * 2);
  bf16_t* Gb1  = (bf16_t*)alloc((size_t)N_TOK * D_HID  * 2);
  bf16_t* Vt1  = (bf16_t*)alloc((size_t)D_HID * N_TOK  * 2);
  bf16_t* Kb2  = (bf16_t*)alloc((size_t)N_TOK * D_HID  * 2);
  bf16_t* Qb2  = (bf16_t*)alloc((size_t)N_TOK * D_HID  * 2);
  bf16_t* Gb2  = (bf16_t*)alloc((size_t)N_TOK * D_HID  * 2);
  bf16_t* Vt2  = (bf16_t*)alloc((size_t)D_HID * N_TOK  * 2);
  float*  O1   = (float*) alloc((size_t)N_TOK * D_HID  * 4);
  float*  O2   = (float*) alloc((size_t)N_TOK * D_HID  * 4);
  float*  part = (float*) alloc(128 * 4);
  float*  sums = (float*) alloc(2 * 4);

  // -- stage 0: precision conversion + weight transpose ---------------------
  {
    int n = N_TOK * D_IN;
    cvt_f32_to_bf16_kernel<<<(n + 255) / 256, 256, 0, stream>>>(x, xb, n);
  }
  {
    int n = N_TOK * D_BOXP;
    cvt_pad_box_kernel<<<(n + 255) / 256, 256, 0, stream>>>(box, boxb);
  }
  {
    int n = D_HID * D_IN;   // 131072
    int g = (n + 255) / 256;
    transpose_w_kernel<<<g, 256, 0, stream>>>(WK1, WtK1, D_IN, D_HID, D_IN);
    transpose_w_kernel<<<g, 256, 0, stream>>>(WQ1, WtQ1, D_IN, D_HID, D_IN);
    transpose_w_kernel<<<g, 256, 0, stream>>>(WV1, WtV1, D_IN, D_HID, D_IN);
    transpose_w_kernel<<<g, 256, 0, stream>>>(WK2, WtK2, D_IN, D_HID, D_IN);
    transpose_w_kernel<<<g, 256, 0, stream>>>(WQ2, WtQ2, D_IN, D_HID, D_IN);
    transpose_w_kernel<<<g, 256, 0, stream>>>(WV2, WtV2, D_IN, D_HID, D_IN);
    int ng = D_HID * D_BOXP;
    transpose_w_kernel<<<(ng + 255) / 256, 256, 0, stream>>>(WG1, WtG1, D_BOX, D_HID, D_BOXP);
    transpose_w_kernel<<<(ng + 255) / 256, 256, 0, stream>>>(WG2, WtG2, D_BOX, D_HID, D_BOXP);
  }

  // -- stage 1: projections (WMMA bf16) -------------------------------------
  dim3 pg(N_TOK / 64, D_HID / 64);   // 64 x 4
  proj_kernel<<<pg, 256, 0, stream>>>(xb, WtK1, Kb1, D_IN, D_IN, D_HID, 0);
  proj_kernel<<<pg, 256, 0, stream>>>(xb, WtQ1, Qb1, D_IN, D_IN, D_HID, 0);
  proj_kernel<<<pg, 256, 0, stream>>>(xb, WtV1, Vt1, D_IN, D_IN, N_TOK, 1);
  proj_kernel<<<pg, 256, 0, stream>>>(xb, WtK2, Kb2, D_IN, D_IN, D_HID, 0);
  proj_kernel<<<pg, 256, 0, stream>>>(xb, WtQ2, Qb2, D_IN, D_IN, D_HID, 0);
  proj_kernel<<<pg, 256, 0, stream>>>(xb, WtV2, Vt2, D_IN, D_IN, N_TOK, 1);
  proj_kernel<<<pg, 256, 0, stream>>>(boxb, WtG1, Gb1, D_BOXP, D_BOXP, D_HID, 0);
  proj_kernel<<<pg, 256, 0, stream>>>(boxb, WtG2, Gb2, D_BOXP, D_BOXP, D_HID, 0);

  // -- stage 2: fused gated-attention (scores never hit HBM) ----------------
  dim3 fg(N_TOK / BM, 2);            // 64 row-blocks x 2 branches
  flash_kernel<<<fg, 256, 0, stream>>>(Kb1, Qb1, Gb1, Vt1, O1,
                                       Kb2, Qb2, Gb2, Vt2, O2, part);

  // -- stage 3: deterministic global sum + epilogue -------------------------
  reduce_sums_kernel<<<1, 32, 0, stream>>>(part, sums);
  {
    int n = N_TOK * D_IN;
    finalize_kernel<<<(n + 255) / 256, 256, 0, stream>>>(O1, O2, sums, x, out);
  }
}
